// MDConv_72215580114961
// MI455X (gfx1250) — matrix operand order repaired and checked by reading, not compiled, not run on hardware
//
#include <hip/hip_runtime.h>

typedef __attribute__((ext_vector_type(16))) __bf16 v16bf;
typedef __attribute__((ext_vector_type(8)))  float  v8f;

#define BATCH 8
#define HH    128
#define WW    128
#define CC    128
#define FF    128
#define LDX   72   // LDS row stride (bf16 elems): 144B/row -> conflict-free b128 reads
#define LDW   72

union Frag { uint4 q[2]; v16bf v; };

__device__ __forceinline__ unsigned pack2bf(float a, float b) {
    union { __bf16 h[2]; unsigned u; } pk;
    pk.h[0] = (__bf16)a; pk.h[1] = (__bf16)b;
    return pk.u;
}

// ---------------------------------------------------------------------------
// Kernel 1: modulate + demodulate, write bf16 weights as wp[b][tap][f][c]
// (c fastest => K-major for the WMMA B fragment loader).
// ---------------------------------------------------------------------------
__global__ __launch_bounds__(128) void modprep(const float* __restrict__ style,
                                               const float* __restrict__ kern,
                                               __bf16* __restrict__ wp)
{
    __shared__ float red[128];
    int bf = blockIdx.x;
    int b  = bf >> 7;
    int f  = bf & 127;
    int c  = threadIdx.x;

    float sm = style[b * CC + c] + 1.0f;
    float t[9];
    float ss = 0.f;
#pragma unroll
    for (int i = 0; i < 9; ++i) {
        float v = kern[((size_t)i * CC + c) * FF + f] * sm;   // (kh,kw,C,F)
        t[i] = v;
        ss += v * v;
    }
    red[c] = ss;
    __syncthreads();
    for (int s = 64; s > 0; s >>= 1) {
        if (c < s) red[c] += red[c + s];
        __syncthreads();
    }
    float dsc = rsqrtf(red[0] + 1e-8f);

    __bf16* dst = wp + (size_t)b * 9 * FF * CC;
#pragma unroll
    for (int i = 0; i < 9; ++i)
        dst[((size_t)i * FF + f) * CC + c] = (__bf16)(t[i] * dsc);
}

// ---------------------------------------------------------------------------
// WMMA fragment loaders (LDS -> VGPR), per ISA 7.12.2 bf16 layouts.
// sX is halo-padded: LDS row index = image w + 1, rows 0 and WW+1 are zeros,
// so the kw shift folds into the address and needs no per-lane select.
// ---------------------------------------------------------------------------
__device__ __forceinline__ v16bf loadA(const __bf16* sx, int pixbase, int kw,
                                       int c0, int lane)
{
    int m  = lane & 15;
    int hh = lane >> 4;
    const __bf16* p = sx + (pixbase + m + kw) * LDX + c0 + 8 * hh;
    Frag fr;
    fr.q[0] = *(const uint4*)(p);        // K = 8h .. 8h+7
    fr.q[1] = *(const uint4*)(p + 16);   // K = 16+8h .. 16+8h+7
    return fr.v;
}

__device__ __forceinline__ v16bf loadB(const __bf16* sw, int fbase, int c0,
                                       int lane)
{
    int n  = lane & 15;
    int hh = lane >> 4;
    const __bf16* p = sw + (fbase + n) * LDW + c0 + 8 * hh;
    Frag fr;
    fr.q[0] = *(const uint4*)(p);
    fr.q[1] = *(const uint4*)(p + 16);
    return fr.v;
}

__device__ __forceinline__ void stage_w(__bf16* dst, const __bf16* __restrict__ wp,
                                        int b, int kh, int kw, int cbase,
                                        int srow, int shalf)
{
    const uint4* gw = (const uint4*)(wp +
        ((((size_t)b * 9 + kh * 3 + kw) * FF + srow) * CC + cbase + 32 * shalf));
    uint4* dw = (uint4*)(dst + srow * LDW + 32 * shalf);
#pragma unroll
    for (int i = 0; i < 4; ++i) dw[i] = gw[i];
}

// ---------------------------------------------------------------------------
// Kernel 2: implicit-GEMM conv. One workgroup per (b, image row h):
// output tile = 128 px x 128 F, K = 9 taps x 128 ch. 8 waves; wave tile =
// 32 px x 64 F = 2x4 accumulators. Weight tiles double-buffered in LDS.
// ---------------------------------------------------------------------------
__global__ __launch_bounds__(256) void modconv(const float* __restrict__ x,
                                               const __bf16* __restrict__ wp,
                                               float* __restrict__ out)
{
    __shared__ __align__(16) __bf16 sX[(WW + 2) * LDX]; // halo-padded input row
    __shared__ __align__(16) __bf16 sW[2][FF * LDW];    // double-buffered tap

    int bh = blockIdx.x;
    int b  = bh >> 7;
    int h  = bh & 127;
    int tid  = threadIdx.x;
    int lane = tid & 31;
    int wv   = tid >> 5;
    int py = (wv >> 1) * 32;
    int fx = (wv & 1) * 64;

    v8f acc[2][4];
#pragma unroll
    for (int pm = 0; pm < 2; ++pm)
#pragma unroll
        for (int fn = 0; fn < 4; ++fn)
            acc[pm][fn] = (v8f){0.f,0.f,0.f,0.f,0.f,0.f,0.f,0.f};

    int srow  = tid >> 1;   // staging row 0..127
    int shalf = tid & 1;    // 32-element half

    for (int it = 0; it < 6; ++it) {      // (kh, channel-half)
        int kh = it >> 1;
        int ch = it & 1;
        int cbase = ch * 64;
        int srch  = h + kh - 1;
        bool valid = (unsigned)srch < (unsigned)HH;

        // ---- stage x row (fp32 -> bf16), halo offset +1 ----
        {
            const float* gx = x + ((((size_t)b * HH + (valid ? srch : 0)) * WW
                                    + srow) * CC + cbase + 32 * shalf);
            __bf16* dx = sX + (srow + 1) * LDX + 32 * shalf;
#pragma unroll
            for (int i = 0; i < 8; ++i) {
                float4 fv = valid ? ((const float4*)gx)[i]
                                  : make_float4(0.f, 0.f, 0.f, 0.f);
                uint2 u;
                u.x = pack2bf(fv.x, fv.y);
                u.y = pack2bf(fv.z, fv.w);
                *(uint2*)(dx + 4 * i) = u;
            }
        }
        // ---- halo columns w=-1 and w=WW: zeros ----
        if (tid < 4) {
            int hw = (tid >> 1) ? (WW + 1) : 0;
            uint4* dz = (uint4*)(sX + hw * LDX + 32 * (tid & 1));
#pragma unroll
            for (int i = 0; i < 4; ++i) dz[i] = make_uint4(0u, 0u, 0u, 0u);
        }
        // ---- stage first weight tile (kw=0) into buffer 0 ----
        stage_w(&sW[0][0], wp, b, kh, 0, cbase, srow, shalf);

        // ---- prefetch next iteration's x row (global_prefetch_b8) ----
        if (it + 1 < 6) {
            int nit = it + 1, nkh = nit >> 1, nch = nit & 1;
            int nsrch = h + nkh - 1;
            if ((unsigned)nsrch < (unsigned)HH) {
                const float* ngx = x + ((((size_t)b * HH + nsrch) * WW + srow) * CC
                                        + nch * 64 + 32 * shalf);
                __builtin_prefetch(ngx, 0, 0);
            }
        }

        __syncthreads();   // staging visible to all waves

        for (int kw = 0; kw < 3; ++kw) {
            // overlap: stage next tap into the other buffer while computing
            if (kw < 2)
                stage_w(&sW[(kw + 1) & 1][0], wp, b, kh, kw + 1, cbase, srow, shalf);

            const __bf16* swb = &sW[kw & 1][0];
#pragma unroll
            for (int c0 = 0; c0 < 64; c0 += 32) {
                v16bf a0 = loadA(sX, py,      kw, c0, lane);
                v16bf a1 = loadA(sX, py + 16, kw, c0, lane);
                v16bf b0 = loadB(swb, fx,      c0, lane);
                v16bf b1 = loadB(swb, fx + 16, c0, lane);
                v16bf b2 = loadB(swb, fx + 32, c0, lane);
                v16bf b3 = loadB(swb, fx + 48, c0, lane);
                acc[0][0] = __builtin_amdgcn_wmma_f32_16x16x32_bf16(false, a0, false, b0, (short)0, acc[0][0], false, false);
                acc[0][1] = __builtin_amdgcn_wmma_f32_16x16x32_bf16(false, a0, false, b1, (short)0, acc[0][1], false, false);
                acc[0][2] = __builtin_amdgcn_wmma_f32_16x16x32_bf16(false, a0, false, b2, (short)0, acc[0][2], false, false);
                acc[0][3] = __builtin_amdgcn_wmma_f32_16x16x32_bf16(false, a0, false, b3, (short)0, acc[0][3], false, false);
                acc[1][0] = __builtin_amdgcn_wmma_f32_16x16x32_bf16(false, a1, false, b0, (short)0, acc[1][0], false, false);
                acc[1][1] = __builtin_amdgcn_wmma_f32_16x16x32_bf16(false, a1, false, b1, (short)0, acc[1][1], false, false);
                acc[1][2] = __builtin_amdgcn_wmma_f32_16x16x32_bf16(false, a1, false, b2, (short)0, acc[1][2], false, false);
                acc[1][3] = __builtin_amdgcn_wmma_f32_16x16x32_bf16(false, a1, false, b3, (short)0, acc[1][3], false, false);
            }
            __syncthreads();   // next-buffer staging visible; guards reuse
        }
    }

    // ---- store: D layout M = r + 8*(lane>=16), N = lane&15 ----
    int h16 = lane >> 4;
    int n   = lane & 15;
    float* ob = out + ((size_t)b * HH + h) * WW * FF;
#pragma unroll
    for (int pm = 0; pm < 2; ++pm) {
#pragma unroll
        for (int fn = 0; fn < 4; ++fn) {
            int fcol = fx + fn * 16 + n;
            int wrow = py + pm * 16 + 8 * h16;
#pragma unroll
            for (int r = 0; r < 8; ++r)
                ob[(size_t)(wrow + r) * FF + fcol] = acc[pm][fn][r];
        }
    }
}

extern "C" void kernel_launch(void* const* d_in, const int* in_sizes, int n_in,
                              void* d_out, int out_size, void* d_ws, size_t ws_size,
                              hipStream_t stream) {
    const float* x     = (const float*)d_in[0];   // (8,128,128,128) fp32
    const float* style = (const float*)d_in[1];   // (8,128) fp32
    const float* kern  = (const float*)d_in[2];   // (3,3,128,128) fp32
    float* out = (float*)d_out;                   // (8,128,128,128) fp32
    __bf16* wp = (__bf16*)d_ws;                   // 8*9*128*128 bf16 = 2.25 MB

    (void)in_sizes; (void)n_in; (void)out_size; (void)ws_size;

    modprep<<<BATCH * FF, 128, 0, stream>>>(style, kern, wp);
    modconv<<<BATCH * HH, 256, 0, stream>>>(x, wp, out);
}
